// SSLPretrainer_1932735284055
// MI455X (gfx1250) — matrix operand order repaired and checked by reading, not compiled, not run on hardware
//
#include <hip/hip_runtime.h>
#include <hip/hip_bf16.h>

typedef unsigned short u16;
typedef unsigned int   u32;
typedef __attribute__((ext_vector_type(16))) __bf16 v16bf;
typedef __attribute__((ext_vector_type(8)))  float  v8f;
typedef __attribute__((ext_vector_type(4)))  u32    u32x4;

struct FragU { u32x4 lo, hi; };

__device__ __forceinline__ u16 f2bf(float f) {
  u32 u = __builtin_bit_cast(u32, f);
  u32 r = (u + 0x7FFFu + ((u >> 16) & 1u)) >> 16;
  return (u16)r;
}

// A fragment: 16x32 bf16 tile from row-major [M,K] (lda in elements).
// Lane L (m=L&15, hi=L>>4) holds K ranges [hi*8, hi*8+8) and [16+hi*8, 16+hi*8+8).
__device__ __forceinline__ v16bf load_a_frag(const u16* __restrict__ A, int lda,
                                             int row0, int k0, int lane) {
  int m = lane & 15, hi = lane >> 4;
  const u16* p = A + (size_t)(row0 + m) * lda + k0 + hi * 8;
  FragU f;
  f.lo = *reinterpret_cast<const u32x4*>(p);
  f.hi = *reinterpret_cast<const u32x4*>(p + 16);
  return __builtin_bit_cast(v16bf, f);
}

// B fragment: 32x16 bf16 tile (K x N) sourced from row-major [N,K] (transposed-B).
// Lane L (n=L&15, kh=L>>4) holds K ranges [kh*8, kh*8+8) and [16+kh*8, 16+kh*8+8) of row n0+n.
__device__ __forceinline__ v16bf load_w_frag(const u16* __restrict__ W, int ldw,
                                             int n0, int k0, int lane, int nmax) {
  int n = n0 + (lane & 15);
  n = n < nmax ? n : nmax;            // clamp for N not multiple of 64 (e.g. N=32)
  int kh = lane >> 4;
  const u16* p = W + (size_t)n * ldw + k0 + kh * 8;
  FragU f;
  f.lo = *reinterpret_cast<const u32x4*>(p);
  f.hi = *reinterpret_cast<const u32x4*>(p + 16);
  return __builtin_bit_cast(v16bf, f);
}

// Generic batched GEMM: C[z] = A[z][M,K] @ W[z][N,K]^T + bias, optional exact GELU,
// f32 and/or bf16 stores. Batch index decomposed as z1=z/zdiv, z2=z%zdiv with
// independent strides so packed-QKV head views work without copies.
__global__ __launch_bounds__(256) void gemm_bf16_wmma(
    const u16* __restrict__ A, int lda, long sA1, long sA2,
    const u16* __restrict__ W, int ldw, long sW1, long sW2,
    const float* __restrict__ bias,
    float* __restrict__ Cf, long sF1, long sF2,
    u16* __restrict__ Cb, long sB1, long sB2,
    int ldc, int M, int N, int K, int zdiv, int act) {
  int z = blockIdx.z;
  int z1 = z / zdiv, z2 = z - z1 * zdiv;
  A += z1 * sA1 + z2 * sA2;
  W += z1 * sW1 + z2 * sW2;
  long offF = z1 * sF1 + z2 * sF2;
  long offB = z1 * sB1 + z2 * sB2;

  int lane = threadIdx.x & 31;
  int wave = threadIdx.x >> 5;
  int row0 = blockIdx.y * 128 + (wave >> 1) * 32;   // 4 waves along M
  int col0 = blockIdx.x * 64 + (wave & 1) * 32;     // 2 waves along N
  if (row0 >= M) return;
  int ncl = N - 1;

  v8f acc00 = {}, acc01 = {}, acc10 = {}, acc11 = {};
  for (int k0 = 0; k0 < K; k0 += 32) {
    v16bf a0 = load_a_frag(A, lda, row0, k0, lane);
    v16bf a1 = load_a_frag(A, lda, row0 + 16, k0, lane);
    v16bf b0 = load_w_frag(W, ldw, col0, k0, lane, ncl);
    v16bf b1 = load_w_frag(W, ldw, col0 + 16, k0, lane, ncl);
    acc00 = __builtin_amdgcn_wmma_f32_16x16x32_bf16(false, a0, false, b0, (short)0, acc00, false, false);
    acc01 = __builtin_amdgcn_wmma_f32_16x16x32_bf16(false, a0, false, b1, (short)0, acc01, false, false);
    acc10 = __builtin_amdgcn_wmma_f32_16x16x32_bf16(false, a1, false, b0, (short)0, acc10, false, false);
    acc11 = __builtin_amdgcn_wmma_f32_16x16x32_bf16(false, a1, false, b1, (short)0, acc11, false, false);
  }

  int nlo = lane & 15, mhi = lane >> 4;
  int nA = col0 + nlo, nB = col0 + 16 + nlo;
  float biasA = 0.f, biasB = 0.f;
  if (bias) {
    biasA = bias[nA < ncl ? nA : ncl];
    biasB = bias[nB < ncl ? nB : ncl];
  }
#pragma unroll
  for (int r = 0; r < 8; r++) {
    int rowA = row0 + mhi * 8 + r;
    int rowB = rowA + 16;
    float vals[4] = {acc00[r] + biasA, acc01[r] + biasB, acc10[r] + biasA, acc11[r] + biasB};
    int   rows[4] = {rowA, rowA, rowB, rowB};
    int   cols[4] = {nA, nB, nA, nB};
#pragma unroll
    for (int q = 0; q < 4; q++) {
      float v = vals[q];
      if (act == 1) v = 0.5f * v * (1.0f + erff(v * 0.70710678118654752f)); // exact GELU
      if (cols[q] < N) {
        long o = (long)rows[q] * ldc + cols[q];
        if (Cf) Cf[offF + o] = v;
        if (Cb) Cb[offB + o] = f2bf(v);
      }
    }
  }
}

__global__ void f32_to_bf16_k(const float* __restrict__ s, u16* __restrict__ d, int n) {
  int i = blockIdx.x * 256 + threadIdx.x;
  if (i < n) d[i] = f2bf(s[i]);
}

// Gather visible patches from image: x[16,3,512,512] -> Pvis bf16 [16*256, 768]
__global__ void patchify_vis_k(const float* __restrict__ x, const int* __restrict__ vis,
                               u16* __restrict__ out) {
  int i = blockIdx.x * 256 + threadIdx.x;      // over 4096*768
  int d = i % 768;
  int t = i / 768;                              // b*256 + j
  int b = t >> 8;
  int p = vis[t];
  int gh = p >> 5, gw = p & 31;
  int c = d >> 8;
  int rem = d & 255;
  int py = rem >> 4, px = rem & 15;
  long src = (((long)(b * 3 + c) * 512) + gh * 16 + py) * 512 + gw * 16 + px;
  out[(long)t * 768 + d] = f2bf(x[src]);
}

// V section of packed QKV -> Vt[b*4+h][32][S]
__global__ void unpack_vt_k(const u16* __restrict__ qkv, u16* __restrict__ vt,
                            int S, int total) {
  int i = blockIdx.x * 256 + threadIdx.x;
  if (i >= total) return;
  int d = i & 127;
  int t = i >> 7;
  int s = t % S, b = t / S;
  int h = d >> 5, dd = d & 31;
  vt[((long)(b * 4 + h) * 32 + dd) * S + s] = qkv[(long)t * 384 + 256 + d];
}

// One wave per row: x = LN(res + delta)*w + b; write f32 + bf16 (D=128)
__global__ __launch_bounds__(256) void add_ln_k(const float* __restrict__ res,
                                                const float* __restrict__ delta,
                                                const float* __restrict__ w,
                                                const float* __restrict__ b,
                                                float* __restrict__ xf,
                                                u16* __restrict__ xb) {
  int row = blockIdx.x * 8 + (threadIdx.x >> 5);
  int lane = threadIdx.x & 31;
  long base = (long)row * 128 + lane * 4;
  float v[4];
#pragma unroll
  for (int j = 0; j < 4; j++) v[j] = res[base + j] + delta[base + j];
  float s = v[0] + v[1] + v[2] + v[3];
  for (int m = 16; m; m >>= 1) s += __shfl_xor(s, m, 32);
  float mean = s * (1.0f / 128.0f);
  float q = 0.f;
#pragma unroll
  for (int j = 0; j < 4; j++) { v[j] -= mean; q += v[j] * v[j]; }
  for (int m = 16; m; m >>= 1) q += __shfl_xor(q, m, 32);
  float inv = rsqrtf(q * (1.0f / 128.0f) + 1e-5f);
#pragma unroll
  for (int j = 0; j < 4; j++) {
    float y = v[j] * inv * w[lane * 4 + j] + b[lane * 4 + j];
    xf[base + j] = y;
    xb[base + j] = f2bf(y);
  }
}

// One wave per row: softmax(scores*scale) f32 -> bf16
__global__ __launch_bounds__(256) void softmax_k(const float* __restrict__ s,
                                                 u16* __restrict__ o, int S, float scale) {
  int row = blockIdx.x * 8 + (threadIdx.x >> 5);
  int lane = threadIdx.x & 31;
  const float* sr = s + (long)row * S;
  float m = -3.0e38f;
  for (int i = lane; i < S; i += 32) m = fmaxf(m, sr[i]);
  for (int x = 16; x; x >>= 1) m = fmaxf(m, __shfl_xor(m, x, 32));
  float sum = 0.f;
  for (int i = lane; i < S; i += 32) sum += expf((sr[i] - m) * scale);
  for (int x = 16; x; x >>= 1) sum += __shfl_xor(sum, x, 32);
  float r = 1.0f / sum;
  u16* orow = o + (long)row * S;
  for (int i = lane; i < S; i += 32) orow[i] = f2bf(expf((sr[i] - m) * scale) * r);
}

// full = mask_token + pos everywhere
__global__ void fill_dec_k(const float* __restrict__ mt, const float* __restrict__ pos,
                           float* __restrict__ xf, u16* __restrict__ xb) {
  int i = blockIdx.x * 256 + threadIdx.x;   // 16*1024*128
  int d = i & 127;
  int p = (i >> 7) & 1023;
  float v = mt[d] + pos[p * 128 + d];
  xf[i] = v;
  xb[i] = f2bf(v);
}

// scatter encoded visible tokens back (overwrites fill), + pos
__global__ void scatter_enc_k(const float* __restrict__ enc, const int* __restrict__ vis,
                              const float* __restrict__ pos,
                              float* __restrict__ xf, u16* __restrict__ xb) {
  int i = blockIdx.x * 256 + threadIdx.x;   // 16*256*128
  int d = i & 127;
  int t = i >> 7;
  int b = t >> 8;
  int p = vis[t];
  float v = enc[(long)t * 128 + d] + pos[p * 128 + d];
  long o = ((long)(b * 1024 + p)) * 128 + d;
  xf[o] = v;
  xb[o] = f2bf(v);
}

__global__ void unpatchify_k(const float* __restrict__ rec, float* __restrict__ out) {
  int i = blockIdx.x * 256 + threadIdx.x;   // 16*3*512*512
  int px = i & 511;
  int t = i >> 9;
  int py = t & 511;
  int u = t >> 9;
  int c = u % 3;
  int b = u / 3;
  int gh = py >> 4, iy = py & 15;
  int gw = px >> 4, ix = px & 15;
  int p = gh * 32 + gw;
  int d = c * 256 + iy * 16 + ix;
  out[i] = rec[((long)(b * 1024 + p)) * 768 + d];
}

// -------------------- host side --------------------
static inline int cdiv(int a, int b) { return (a + b - 1) / b; }

static void gemm(hipStream_t st,
                 const u16* A, int lda, long sA1, long sA2,
                 const u16* W, int ldw, long sW1, long sW2,
                 const float* bias,
                 float* Cf, long f1, long f2,
                 u16* Cb, long b1, long b2,
                 int ldc, int M, int N, int K, int nz, int zdiv, int act) {
  dim3 g(cdiv(N, 64), cdiv(M, 128), nz);
  gemm_bf16_wmma<<<g, dim3(256), 0, st>>>(A, lda, sA1, sA2, W, ldw, sW1, sW2, bias,
                                          Cf, f1, f2, Cb, b1, b2, ldc, M, N, K, zdiv, act);
}

static void run_stack(hipStream_t st, int L, int S, int Bt,
                      float* xf, u16* xb,
                      const u16* wqkv, const float* bqkv,
                      const u16* wout, const float* bout,
                      const float* ln1w, const float* ln1b,
                      const u16* wff1, const float* bff1,
                      const u16* wff2, const float* bff2,
                      const float* ln2w, const float* ln2b,
                      u16* qkvb, u16* vt, float* sbuf, u16* attb,
                      u16* attnc, u16* hbuf, float* t2) {
  const int D = 128, NH = 4, HD = 32, FF = 512, CH = 8;
  int T = Bt * S;
  int nbh = Bt * NH;
  const float scale = 0.17677669529663687f;   // 1/sqrt(32)
  for (int l = 0; l < L; l++) {
    const u16* wq = wqkv + (size_t)l * 3 * D * D;
    const u16* wo = wout + (size_t)l * D * D;
    const u16* w1 = wff1 + (size_t)l * FF * D;
    const u16* w2 = wff2 + (size_t)l * D * FF;
    // QKV projection -> bf16 [T, 384]
    gemm(st, xb, D, 0, 0, wq, D, 0, 0, bqkv + l * 3 * D,
         nullptr, 0, 0, qkvb, 0, 0, 3 * D, T, 3 * D, D, 1, 1, 0);
    // V -> Vt
    unpack_vt_k<<<cdiv(T * D, 256), 256, 0, st>>>(qkvb, vt, S, T * D);
    // attention, chunks of CH head-batches to bound scores scratch
    for (int g = 0; g < nbh / CH; g++) {
      int z0 = g * CH;
      int b0 = z0 / NH;                       // CH multiple of NH -> head offset 0
      const u16* qbase = qkvb + (size_t)b0 * S * 3 * D;
      const u16* kbase = qbase + D;
      // scores = Q @ K^T  (M=S, N=S, K=32), per (b,h)
      gemm(st, qbase, 3 * D, (long)S * 3 * D, 32,
           kbase, 3 * D, (long)S * 3 * D, 32,
           nullptr,
           sbuf, (long)NH * S * S, (long)S * S,
           nullptr, 0, 0,
           S, S, S, HD, CH, NH, 0);
      // softmax rows -> bf16 att
      softmax_k<<<(CH * S) / 8, 256, 0, st>>>(sbuf, attb, S, scale);
      // out_h = att @ V = att @ Vt^T  (M=S, N=32, K=S), concat heads into attnc
      gemm(st, attb, S, (long)NH * S * S, (long)S * S,
           vt + (size_t)z0 * HD * S, S, (long)NH * HD * S, (long)HD * S,
           nullptr,
           nullptr, 0, 0,
           attnc + (size_t)b0 * S * D, (long)S * D, 32,
           D, S, HD, S, CH, NH, 0);
    }
    // output projection
    gemm(st, attnc, D, 0, 0, wo, D, 0, 0, bout + l * D,
         t2, 0, 0, nullptr, 0, 0, D, T, D, D, 1, 1, 0);
    add_ln_k<<<T / 8, 256, 0, st>>>(xf, t2, ln1w + l * D, ln1b + l * D, xf, xb);
    // FF1 + exact GELU -> bf16
    gemm(st, xb, D, 0, 0, w1, D, 0, 0, bff1 + l * FF,
         nullptr, 0, 0, hbuf, 0, 0, FF, T, FF, D, 1, 1, 1);
    // FF2
    gemm(st, hbuf, FF, 0, 0, w2, FF, 0, 0, bff2 + l * D,
         t2, 0, 0, nullptr, 0, 0, D, T, D, FF, 1, 1, 0);
    add_ln_k<<<T / 8, 256, 0, st>>>(xf, t2, ln2w + l * D, ln2b + l * D, xf, xb);
  }
}

extern "C" void kernel_launch(void* const* d_in, const int* in_sizes, int n_in,
                              void* d_out, int out_size, void* d_ws, size_t ws_size,
                              hipStream_t stream) {
  (void)in_sizes; (void)n_in; (void)out_size; (void)ws_size;
  const float* x        = (const float*)d_in[0];
  const int*   vis      = (const int*)d_in[1];
  const float* w_patch  = (const float*)d_in[2];
  const float* b_patch  = (const float*)d_in[3];
  const float* mask_tok = (const float*)d_in[4];
  const float* pos      = (const float*)d_in[5];
  const float* w_recon  = (const float*)d_in[6];
  const float* b_recon  = (const float*)d_in[7];
  const float* e_qkv_w = (const float*)d_in[8];
  const float* e_qkv_b = (const float*)d_in[9];
  const float* e_out_w = (const float*)d_in[10];
  const float* e_out_b = (const float*)d_in[11];
  const float* e_ln1_w = (const float*)d_in[12];
  const float* e_ln1_b = (const float*)d_in[13];
  const float* e_ff1_w = (const float*)d_in[14];
  const float* e_ff1_b = (const float*)d_in[15];
  const float* e_ff2_w = (const float*)d_in[16];
  const float* e_ff2_b = (const float*)d_in[17];
  const float* e_ln2_w = (const float*)d_in[18];
  const float* e_ln2_b = (const float*)d_in[19];
  const float* g_qkv_w = (const float*)d_in[20];
  const float* g_qkv_b = (const float*)d_in[21];
  const float* g_out_w = (const float*)d_in[22];
  const float* g_out_b = (const float*)d_in[23];
  const float* g_ln1_w = (const float*)d_in[24];
  const float* g_ln1_b = (const float*)d_in[25];
  const float* g_ff1_w = (const float*)d_in[26];
  const float* g_ff1_b = (const float*)d_in[27];
  const float* g_ff2_w = (const float*)d_in[28];
  const float* g_ff2_b = (const float*)d_in[29];
  const float* g_ln2_w = (const float*)d_in[30];
  const float* g_ln2_b = (const float*)d_in[31];

  // bump allocator over workspace (256B aligned)
  char* wsb = (char*)d_ws;
  size_t off = 0;
  auto take = [&](size_t bytes) -> void* {
    off = (off + 255) & ~(size_t)255;
    void* p = wsb + off;
    off += bytes;
    return p;
  };

  u16* wb_patch = (u16*)take((size_t)128 * 768 * 2);
  u16* wb_eqkv  = (u16*)take((size_t)6 * 384 * 128 * 2);
  u16* wb_eout  = (u16*)take((size_t)6 * 128 * 128 * 2);
  u16* wb_eff1  = (u16*)take((size_t)6 * 512 * 128 * 2);
  u16* wb_eff2  = (u16*)take((size_t)6 * 128 * 512 * 2);
  u16* wb_dqkv  = (u16*)take((size_t)2 * 384 * 128 * 2);
  u16* wb_dout  = (u16*)take((size_t)2 * 128 * 128 * 2);
  u16* wb_dff1  = (u16*)take((size_t)2 * 512 * 128 * 2);
  u16* wb_dff2  = (u16*)take((size_t)2 * 128 * 512 * 2);
  u16* wb_recon = (u16*)take((size_t)768 * 128 * 2);
  u16* pvis     = (u16*)take((size_t)4096 * 768 * 2);
  float* xf_e   = (float*)take((size_t)4096 * 128 * 4);
  u16* xb_e     = (u16*)take((size_t)4096 * 128 * 2);
  float* xf_d   = (float*)take((size_t)16384 * 128 * 4);
  u16* xb_d     = (u16*)take((size_t)16384 * 128 * 2);
  u16* qkvb     = (u16*)take((size_t)16384 * 384 * 2);
  u16* vt       = (u16*)take((size_t)64 * 32 * 1024 * 2);
  float* sbuf   = (float*)take((size_t)8 * 1024 * 1024 * 4);
  u16* attb     = (u16*)take((size_t)8 * 1024 * 1024 * 2);
  u16* attnc    = (u16*)take((size_t)16384 * 128 * 2);
  u16* hbuf     = (u16*)take((size_t)16384 * 512 * 2);
  float* t2     = (float*)take((size_t)16384 * 128 * 4);
  float* rec    = (float*)take((size_t)16384 * 768 * 4);

  auto conv = [&](const float* s, u16* d, int n) {
    f32_to_bf16_k<<<cdiv(n, 256), 256, 0, stream>>>(s, d, n);
  };
  conv(w_patch, wb_patch, 128 * 768);
  conv(e_qkv_w, wb_eqkv, 6 * 384 * 128);
  conv(e_out_w, wb_eout, 6 * 128 * 128);
  conv(e_ff1_w, wb_eff1, 6 * 512 * 128);
  conv(e_ff2_w, wb_eff2, 6 * 128 * 512);
  conv(g_qkv_w, wb_dqkv, 2 * 384 * 128);
  conv(g_out_w, wb_dout, 2 * 128 * 128);
  conv(g_ff1_w, wb_dff1, 2 * 512 * 128);
  conv(g_ff2_w, wb_dff2, 2 * 128 * 512);
  conv(w_recon, wb_recon, 768 * 128);

  // gather visible patches, convert to bf16
  patchify_vis_k<<<(4096 * 768) / 256, 256, 0, stream>>>(x, vis, pvis);

  // patch embedding: [4096,768] @ [128,768]^T + b -> f32 + bf16
  gemm(stream, pvis, 768, 0, 0, wb_patch, 768, 0, 0, b_patch,
       xf_e, 0, 0, xb_e, 0, 0, 128, 4096, 128, 768, 1, 1, 0);

  // encoder: 6 layers, S=256
  run_stack(stream, 6, 256, 16, xf_e, xb_e,
            wb_eqkv, e_qkv_b, wb_eout, e_out_b, e_ln1_w, e_ln1_b,
            wb_eff1, e_ff1_b, wb_eff2, e_ff2_b, e_ln2_w, e_ln2_b,
            qkvb, vt, sbuf, attb, attnc, hbuf, t2);

  // decoder input: mask_token + pos everywhere, scatter encoded tokens (+pos)
  fill_dec_k<<<(16 * 1024 * 128) / 256, 256, 0, stream>>>(mask_tok, pos, xf_d, xb_d);
  scatter_enc_k<<<(16 * 256 * 128) / 256, 256, 0, stream>>>(xf_e, vis, pos, xf_d, xb_d);

  // decoder: 2 layers, S=1024
  run_stack(stream, 2, 1024, 16, xf_d, xb_d,
            wb_dqkv, g_qkv_b, wb_dout, g_out_b, g_ln1_w, g_ln1_b,
            wb_dff1, g_ff1_b, wb_dff2, g_ff2_b, g_ln2_w, g_ln2_b,
            qkvb, vt, sbuf, attb, attnc, hbuf, t2);

  // reconstruction: [16384,128] @ [768,128]^T + b -> f32
  gemm(stream, xb_d, 128, 0, 0, wb_recon, 128, 0, 0, b_recon,
       rec, 0, 0, nullptr, 0, 0, 768, 16384, 768, 128, 1, 1, 0);

  // unpatchify -> d_out [16,3,512,512] f32
  unpatchify_k<<<(16 * 3 * 512 * 512) / 256, 256, 0, stream>>>(rec, (float*)d_out);
}
